// ZoeDepthMetricDepthEstimationHead_42167988912419
// MI455X (gfx1250) — matrix operand order repaired and checked by reading, not compiled, not run on hardware
//
#include <hip/hip_runtime.h>

typedef __attribute__((ext_vector_type(16))) _Float16 v16h;
typedef __attribute__((ext_vector_type(8)))  float    v8f;
typedef __attribute__((ext_vector_type(4)))  float    f32x4;

#define ALPHA  300.0f
#define NATT   16
#define BB     4
#define CC     128
#define HH     192
#define WW     192
#define HS     96
#define WS     96
#define NBINS  64
#define PTILE  256
#define CPAD   136          // f16 pitch (68 dwords -> lane bank stride 4 mod 64)
#define APITCH 17           // att f32 pitch

// ---- LDS fragment loaders (16B-aligned b128 pairs) ----
// A (16x32 f16, MxK): lane L holds row M=L&15; halves [k0..k0+7] and [k0+16..k0+23],
// k0 = ks*32 + (L>>4)*8
__device__ __forceinline__ v16h lds_loadA(const _Float16* row, int k0) {
  union { v16h v; f32x4 q[2]; } u;
  u.q[0] = *(const f32x4*)(row + k0);
  u.q[1] = *(const f32x4*)(row + k0 + 16);
  return u.v;
}
// B (32x16 f16, KxN): lane L holds col N=L&15; halves [k0..k0+7] and [k0+16..k0+23],
// k0 = ks*32 + (L>>4)*8
__device__ __forceinline__ v16h lds_loadB(const _Float16* row, int k0) {
  union { v16h v; f32x4 q[2]; } u;
  u.q[0] = *(const f32x4*)(row + k0);
  u.q[1] = *(const f32x4*)(row + k0 + 16);
  return u.v;
}

extern "C" __global__ __launch_bounds__(256)
void zoe_head_fused(const float* __restrict__ x,
                    const float* __restrict__ prev_bin,
                    const float* __restrict__ pbe,
                    const float* __restrict__ w1,
                    const float* __restrict__ b1,
                    const float* __restrict__ w2,
                    const float* __restrict__ b2,
                    float* __restrict__ out0,
                    float* __restrict__ out1)
{
  extern __shared__ char smem[];
  _Float16* hL  = (_Float16*)smem;            // [256][CPAD] h  (pixel-major)
  _Float16* h1L = hL  + PTILE * CPAD;         // [256][CPAD] h1 (pixel-major)
  _Float16* w1L = h1L + PTILE * CPAD;         // [128][CPAD]
  _Float16* w2L = w1L + CC    * CPAD;         // [16][CPAD]  (even rows of w2)
  float*    b1L = (float*)(w2L + NATT*CPAD);  // [128]
  float*    b2L = b1L + CC;                   // [16]
  float*   attL = b2L + NATT;                 // [256][APITCH]

  const int t  = threadIdx.x;
  const int wg = blockIdx.x;

  // ---------- stage 0: weights/bias -> LDS (f32 -> f16) ----------
  for (int i = t; i < CC*CC; i += 256) {
    int o = i >> 7, c = i & 127;
    w1L[o*CPAD + c] = (_Float16)w1[i];
  }
  for (int i = t; i < NATT*CC; i += 256) {
    int o2 = i >> 7, c = i & 127;
    w2L[o2*CPAD + c] = (_Float16)w2[(2*o2)*CC + c];   // even channels only
  }
  if (t < CC)   b1L[t] = b1[t];
  if (t < NATT) b2L[t] = b2[2*t];

  // ---------- stage 1: h = x + bilinear_up(pbe) -> LDS f16 ----------
  const int p   = wg * PTILE + t;
  const int b   = p / (HH*WW);
  const int rem = p - b*(HH*WW);
  const int y   = rem / WW;
  const int xw  = rem - y*WW;

  const float sc = (float)(HS-1) / (float)(HH-1);   // align_corners scale
  float fy = y  * sc; int y0 = (int)fy; float wy = fy - (float)y0;
  float fx = xw * sc; int x0 = (int)fx; float wx = fx - (float)x0;
  int y1 = (y0 + 1 < HS) ? y0 + 1 : HS - 1;
  int x1 = (x0 + 1 < WS) ? x0 + 1 : WS - 1;
  const int o00 = y0*WS + x0, o01 = y0*WS + x1, o10 = y1*WS + x0, o11 = y1*WS + x1;
  const float w00 = (1.f-wy)*(1.f-wx), w01 = (1.f-wy)*wx,
              w10 = wy*(1.f-wx),       w11 = wy*wx;

  const float* xb = x   + (size_t)b*CC*HH*WW + (size_t)y*WW + xw;
  const float* pb = pbe + (size_t)b*CC*HS*WS;
  for (int c = 0; c < CC; ++c) {
    const float* q = pb + (size_t)c*HS*WS;
    float pe = q[o00]*w00 + q[o01]*w01 + q[o10]*w10 + q[o11]*w11;
    hL[t*CPAD + c] = (_Float16)(xb[(size_t)c*HH*WW] + pe);
  }
  __syncthreads();

  // ---------- stage 2: GEMM1  h1 = relu(W1 h + b1)  via WMMA ----------
  const int wave = t >> 5;
  const int lane = t & 31;
  const int hi   = lane >> 4;
  const int ln   = lane & 15;

  for (int tile = wave; tile < 128; tile += 8) {     // 8 O-tiles x 16 P-tiles
    const int otile = tile & 7;
    const int ptile = tile >> 3;
    v8f acc = {};
    const _Float16* arow = w1L + (otile*16 + ln)*CPAD;
    const _Float16* brow = hL  + (ptile*16 + ln)*CPAD;
#pragma unroll
    for (int ks = 0; ks < 4; ++ks) {
      v16h a  = lds_loadA(arow, ks*32 + hi*8);
      v16h bm = lds_loadB(brow, ks*32 + hi*8);
      acc = __builtin_amdgcn_wmma_f32_16x16x32_f16(false, a, false, bm,
                                                   (short)0, acc, false, false);
    }
    const int obase = otile*16 + hi*8;       // D: M = r + 8*hi
    const int prow  = ptile*16 + ln;         // D: N = ln
    union { f32x4 q; _Float16 hh[8]; } pk;
#pragma unroll
    for (int r = 0; r < 8; ++r) {
      float v = acc[r] + b1L[obase + r];
      pk.hh[r] = (_Float16)fmaxf(v, 0.0f);
    }
    *(f32x4*)(h1L + prow*CPAD + obase) = pk.q;   // one ds_store_b128
  }
  __syncthreads();

  // ---------- stage 3: GEMM2  att = relu(W2e h1 + b2e) + 1e-3 ----------
#pragma unroll
  for (int pt = 0; pt < 2; ++pt) {
    const int ptile = wave*2 + pt;
    v8f acc = {};
    const _Float16* arow = w2L + ln*CPAD;
    const _Float16* brow = h1L + (ptile*16 + ln)*CPAD;
#pragma unroll
    for (int ks = 0; ks < 4; ++ks) {
      v16h a  = lds_loadA(arow, ks*32 + hi*8);
      v16h bm = lds_loadB(brow, ks*32 + hi*8);
      acc = __builtin_amdgcn_wmma_f32_16x16x32_f16(false, a, false, bm,
                                                   (short)0, acc, false, false);
    }
    const int prow = ptile*16 + ln;
#pragma unroll
    for (int r = 0; r < 8; ++r) {
      const int o2 = hi*8 + r;
      attL[prow*APITCH + o2] = fmaxf(acc[r] + b2L[o2], 0.0f) + 0.001f;
    }
  }
  __syncthreads();

  // ---------- stage 4: attractors + bitonic-64 sort (per pixel) ----------
  float att[NATT];
#pragma unroll
  for (int i = 0; i < NATT; ++i) att[i] = attL[t*APITCH + i];

  const float* pbb = prev_bin + (size_t)b*NBINS*HS*WS;
  const size_t outOff = (size_t)b*NBINS*HH*WW + (size_t)y*WW + xw;
  float v[NBINS];
#pragma unroll
  for (int j = 0; j < NBINS; ++j) {
    const float* q = pbb + (size_t)j*HS*WS;
    float bc = q[o00]*w00 + q[o01]*w01 + q[o10]*w10 + q[o11]*w11;
    float s = 0.f;
#pragma unroll
    for (int i = 0; i < NATT; ++i) {
      float dx = att[i] - bc;
      s += dx / fmaf(ALPHA*dx, dx, 1.0f);
    }
    float bnc = bc + s * (1.0f/16.0f);
    out0[outOff + (size_t)j*HH*WW] = bnc;            // bin_new_centers
    v[j] = fmaf(10.0f - 0.001f, bnc, 0.001f);        // pre-sort bin_centers
  }

  // in-register bitonic sort, ascending (all indices compile-time constants)
#pragma unroll
  for (int k = 2; k <= NBINS; k <<= 1) {
#pragma unroll
    for (int jj = k >> 1; jj > 0; jj >>= 1) {
#pragma unroll
      for (int i = 0; i < NBINS; ++i) {
        const int l = i ^ jj;
        if (l > i) {
          const bool up = ((i & k) == 0);
          float a0 = v[i], a1 = v[l];
          float lo = fminf(a0, a1), hg = fmaxf(a0, a1);
          v[i] = up ? lo : hg;
          v[l] = up ? hg : lo;
        }
      }
    }
  }

#pragma unroll
  for (int j = 0; j < NBINS; ++j)
    out1[outOff + (size_t)j*HH*WW] = fminf(fmaxf(v[j], 0.001f), 10.0f);
}

extern "C" void kernel_launch(void* const* d_in, const int* in_sizes, int n_in,
                              void* d_out, int out_size, void* d_ws, size_t ws_size,
                              hipStream_t stream) {
  const float* x        = (const float*)d_in[0];
  const float* prev_bin = (const float*)d_in[1];
  const float* pbe      = (const float*)d_in[2];
  const float* w1       = (const float*)d_in[3];
  const float* b1       = (const float*)d_in[4];
  const float* w2       = (const float*)d_in[5];
  const float* b2       = (const float*)d_in[6];
  float* out0 = (float*)d_out;
  float* out1 = out0 + (size_t)BB*NBINS*HH*WW;   // tuple element 2

  const int npix   = BB*HH*WW;                   // 147456
  const int blocks = npix / PTILE;               // 576
  const size_t lds = (size_t)(2*PTILE*CPAD + CC*CPAD + NATT*CPAD) * sizeof(_Float16)
                   + (size_t)(CC + NATT + PTILE*APITCH) * sizeof(float); // ~192 KB < 320 KB/WGP

  zoe_head_fused<<<blocks, 256, lds, stream>>>(x, prev_bin, pbe, w1, b1, w2, b2,
                                               out0, out1);
}
